// Attention_54125177864194
// MI455X (gfx1250) — matrix operand order repaired and checked by reading, not compiled, not run on hardware
//
#include <hip/hip_runtime.h>

// Problem constants (match reference)
#define Bn 8
#define Sn 4096
#define En 2048
#define Dn 128

typedef __attribute__((ext_vector_type(16))) __bf16 v16bf;
typedef __attribute__((ext_vector_type(8)))  __bf16 v8bf;
typedef __attribute__((ext_vector_type(8)))  float  v8f;

static __device__ __forceinline__ v16bf cat8(v8bf lo, v8bf hi) {
  return __builtin_shufflevector(lo, hi, 0,1,2,3,4,5,6,7,8,9,10,11,12,13,14,15);
}

static __device__ __forceinline__ v8f wmma_bf16(v16bf a, v16bf b, v8f c) {
  // D = A(16x32 bf16) * B(32x16 bf16) + C(16x16 f32)
  return __builtin_amdgcn_wmma_f32_16x16x32_bf16(false, a, false, b, (short)0, c, false, false);
}

// Hard scheduling fence: nothing crosses (keeps load-issue blocks ahead of
// the WMMA blocks so waits decay instead of hitting loadcnt==0 every time).
static __device__ __forceinline__ void sched_fence() {
  __builtin_amdgcn_sched_barrier(0);
}

// ---------------------------------------------------------------------------
// Kernel 0: pack Wq|Wk|Wv (each [E][128] f32, row-major) into transposed bf16
// panel Wt[384][2048] so B-fragments for the projection GEMM are contiguous.
// ---------------------------------------------------------------------------
__global__ void prep_weights(const float* __restrict__ Wq,
                             const float* __restrict__ Wk,
                             const float* __restrict__ Wv,
                             __bf16* __restrict__ Wt) {
  int idx = blockIdx.x * blockDim.x + threadIdx.x;   // 384*2048 threads
  int c = idx >> 11;          // column in [0,384)
  int e = idx & (En - 1);     // row in [0,2048)
  float v;
  if (c < 128)       v = Wq[(size_t)e * Dn + c];
  else if (c < 256)  v = Wk[(size_t)e * Dn + (c - 128)];
  else               v = Wv[(size_t)e * Dn + (c - 256)];
  Wt[(size_t)c * En + e] = (__bf16)v;
}

// ---------------------------------------------------------------------------
// Kernel 1: QKV projection. One wave computes a 16-row x 128-col tile of one
// of Q/K/V (which = task%3). x converted f32->bf16 on the fly; outputs bf16:
// Q,K row-major [B*S][128]; V transposed Vt[B][128][S].
// Pipeline per k-step: [issue 16 B-frag loads + 4 next-A loads] | fence |
// [convert previous A, 8 WMMAs].
// ---------------------------------------------------------------------------
__global__ __launch_bounds__(128) void qkv_proj(
    const float*  __restrict__ x,  const __bf16* __restrict__ Wt,
    const float*  __restrict__ bq, const float*  __restrict__ bk,
    const float*  __restrict__ bv,
    __bf16* __restrict__ Qb, __bf16* __restrict__ Kb, __bf16* __restrict__ Vt)
{
  const int lane = threadIdx.x & 31;
  const int wid  = threadIdx.x >> 5;
  const int task = blockIdx.x * 4 + wid;   // 6144 tasks
  const int rowblk = task / 3;             // 0..2047 (16-row blocks of B*S)
  const int which  = task % 3;             // 0=Q 1=K 2=V
  const int hi = lane >> 4;
  const int ln = lane & 15;

  const size_t row0 = (size_t)rowblk * 16;
  const float*  xrow  = x  + (row0 + (size_t)ln) * En;        // A row for lane
  const __bf16* wbase = Wt + ((size_t)which * 128 + ln) * En; // B row for lane

  v8f acc[8];
  #pragma unroll
  for (int t = 0; t < 8; t++) { v8f z = {0,0,0,0,0,0,0,0}; acc[t] = z; }

  // Prime the A pipeline (k0 = 0).
  v8f alo = *(const v8f*)(xrow + hi * 8);
  v8f ahi = *(const v8f*)(xrow + 16 + hi * 8);

  const __bf16* wptr = wbase + hi * 8;      // advances 32 per k-step
  const float*  xnxt = xrow + hi * 8;       // advances 32 per k-step

  for (int k0 = 0; k0 < En; k0 += 32) {
    // ---- load-issue block ----
    v16bf bf[8];
    #pragma unroll
    for (int t = 0; t < 8; t++)
      bf[t] = cat8(*(const v8bf*)(wptr + (size_t)t * 16 * En), *(const v8bf*)(wptr + (size_t)t * 16 * En + 16));

    // Next A chunk (wrap on last iteration; harmless reload of k=0).
    const float* xn = (k0 + 32 < En) ? (xnxt + 32) : (xrow + hi * 8);
    v8f nlo = *(const v8f*)(xn);
    v8f nhi = *(const v8f*)(xn + 16);

    sched_fence();

    // ---- compute block ----
    v16bf a;
    #pragma unroll
    for (int i = 0; i < 8; i++) { a[i] = (__bf16)alo[i]; a[8 + i] = (__bf16)ahi[i]; }
    #pragma unroll
    for (int t = 0; t < 8; t++)
      acc[t] = wmma_bf16(a, bf[t], acc[t]);

    alo = nlo; ahi = nhi;
    wptr += 32; xnxt = xn;
  }

  const float* bias = (which == 0) ? bq : (which == 1) ? bk : bv;
  if (which < 2) {
    __bf16* dst = (which == 0) ? Qb : Kb;
    #pragma unroll
    for (int t = 0; t < 8; t++) {
      float bb = bias[t * 16 + ln];
      #pragma unroll
      for (int j = 0; j < 8; j++) {
        size_t r = row0 + j + hi * 8;
        dst[r * Dn + t * 16 + ln] = (__bf16)(acc[t][j] + bb);
      }
    }
  } else {
    // V transposed: Vt[(b*128 + d)*S + s]; pack 8 consecutive s into one 16B store.
    int b  = (int)(row0 / Sn);
    int s0 = (int)(row0 % Sn);
    #pragma unroll
    for (int t = 0; t < 8; t++) {
      float bb = bias[t * 16 + ln];
      v8bf pk;
      #pragma unroll
      for (int j = 0; j < 8; j++) pk[j] = (__bf16)(acc[t][j] + bb);
      __bf16* dst = Vt + ((size_t)b * Dn + t * 16 + ln) * Sn + s0 + hi * 8;
      *(v8bf*)dst = pk;
    }
  }
}

// ---------------------------------------------------------------------------
// Kernel 2: flash attention. One wave owns a 16-query tile; loops over keys
// in chunks of 32 with online softmax. Pipeline per chunk:
// [issue next-K (8 frags) + current-V (8 frags) loads + L2 prefetch] | fence |
// [score WMMAs on resident K, softmax, LDS re-stripe, PV WMMAs].
// ---------------------------------------------------------------------------
__global__ __launch_bounds__(128) void flash_attn(
    const __bf16* __restrict__ Qb, const __bf16* __restrict__ Kb,
    const __bf16* __restrict__ Vt, float* __restrict__ out)
{
  // Per-wave P staging tile; row stride 40 elements (80B) keeps b128 reads
  // conflict-light and 16B-aligned.
  __shared__ __attribute__((aligned(16))) __bf16 pshare[4][16][40];

  const int lane = threadIdx.x & 31;
  const int wid  = threadIdx.x >> 5;
  const int tile = blockIdx.x * 4 + wid;   // 0..2047
  const int b  = tile >> 8;                // / (S/16)
  const int s0 = (tile & 255) << 4;
  const int hi = lane >> 4;
  const int ln = lane & 15;

  // Q A-fragments for the whole D=128 contraction (4 k-steps), held in regs.
  const __bf16* qrow = Qb + ((size_t)b * Sn + s0 + ln) * Dn;
  v16bf qa[4];
  #pragma unroll
  for (int kk = 0; kk < 4; kk++) {
    v8bf lo = *(const v8bf*)(qrow + kk * 32 + hi * 16);
    v8bf h8 = *(const v8bf*)(qrow + kk * 32 + hi * 16 + 8);
    qa[kk] = cat8(lo, h8);
  }

  v8f acc[8];
  #pragma unroll
  for (int t = 0; t < 8; t++) { v8f z = {0,0,0,0,0,0,0,0}; acc[t] = z; }
  float mst[8], lst[8];
  #pragma unroll
  for (int j = 0; j < 8; j++) { mst[j] = -1e30f; lst[j] = 0.f; }

  const __bf16* kbase = Kb + (size_t)b * Sn * Dn;
  const __bf16* vbase = Vt + (size_t)b * Dn * Sn;

  // K-fragment loader from a row pointer (key row ln of the chunk):
  // frags [0..3] = keys +0..15 (k-steps 0..3), frags [4..7] = keys +16..31.
  auto loadKfrom = [&](v16bf* kf, const __bf16* kr) {
    #pragma unroll
    for (int kk = 0; kk < 4; kk++) {
      kf[kk]     = *(const v16bf*)(kr + kk * 32 + hi * 16);
      kf[4 + kk] = *(const v16bf*)(kr + (size_t)16 * Dn + kk * 32 + hi * 16);
    }
  };

  const __bf16* krow0 = kbase + (size_t)ln * Dn;        // chunk K row ptr
  const __bf16* vrow  = vbase + (size_t)ln * Sn + hi * 16;  // chunk V row ptr

  // Prime the K pipeline.
  v16bf kf[8];
  loadKfrom(kf, krow0);
  const __bf16* krow = krow0;

  for (int key0 = 0; key0 < Sn; key0 += 32) {
    // ---- load-issue block ----
    // Next chunk's K frags (branchless wrap keeps addresses in-bounds).
    const __bf16* nkrow = (key0 + 32 < Sn) ? (krow + (size_t)32 * Dn) : krow0;
    v16bf kn[8];
    loadKfrom(kn, nkrow);

    // This chunk's V frags.
    v16bf vfr[8];
    #pragma unroll
    for (int t = 0; t < 8; t++)
      vfr[t] = *(const v16bf*)(vrow + (size_t)t * 16 * Sn);

    // Warm L2 two chunks ahead (speculative; OOB silently dropped).
    __builtin_prefetch(krow + (size_t)64 * Dn, 0, 3);
    __builtin_prefetch(vrow + 64, 0, 3);

    sched_fence();

    // ---- compute block ----
    // scores tile 16x32 = Q(16x128) @ K^T, K frags resident from last iter.
    v8f sc0 = {0,0,0,0,0,0,0,0}, sc1 = {0,0,0,0,0,0,0,0};
    #pragma unroll
    for (int kk = 0; kk < 4; kk++) sc0 = wmma_bf16(qa[kk], kf[kk], sc0);
    #pragma unroll
    for (int kk = 0; kk < 4; kk++) sc1 = wmma_bf16(qa[kk], kf[4 + kk], sc1);

    // Online softmax. Row r = j + hi*8 lives in VGPR j within each lane-half;
    // xor-shuffles with masks 1/2/4/8 reduce across the 16 lanes of a half.
    #pragma unroll
    for (int j = 0; j < 8; j++) {
      float cm = fmaxf(sc0[j], sc1[j]);
      cm = fmaxf(cm, __shfl_xor(cm, 1));
      cm = fmaxf(cm, __shfl_xor(cm, 2));
      cm = fmaxf(cm, __shfl_xor(cm, 4));
      cm = fmaxf(cm, __shfl_xor(cm, 8));
      float mnew   = fmaxf(mst[j], cm);
      float scalef = __expf(mst[j] - mnew);
      float p0 = __expf(sc0[j] - mnew);
      float p1 = __expf(sc1[j] - mnew);
      float rs = p0 + p1;
      rs += __shfl_xor(rs, 1);
      rs += __shfl_xor(rs, 2);
      rs += __shfl_xor(rs, 4);
      rs += __shfl_xor(rs, 8);
      lst[j] = lst[j] * scalef + rs;
      mst[j] = mnew;
      #pragma unroll
      for (int t = 0; t < 8; t++) acc[t][j] *= scalef;
      int prow = j + hi * 8;
      pshare[wid][prow][ln]      = (__bf16)p0;
      pshare[wid][prow][16 + ln] = (__bf16)p1;
    }

    // Re-stripe P (C/D layout -> A layout) through LDS; LDS ops from one wave
    // are in-order, so no barrier needed (private per-wave tile).
    v8bf plo = *(const v8bf*)&pshare[wid][ln][hi * 16];
    v8bf phi = *(const v8bf*)&pshare[wid][ln][hi * 16 + 8];
    v16bf pa = cat8(plo, phi);

    // acc += P(16x32) @ V(32x128), 8 N-tiles over D; V loads were issued at
    // the top of this iteration, covered by score WMMAs + softmax.
    #pragma unroll
    for (int t = 0; t < 8; t++)
      acc[t] = wmma_bf16(pa, vfr[t], acc[t]);

    // Advance pipeline.
    #pragma unroll
    for (int i = 0; i < 8; i++) kf[i] = kn[i];
    krow = nkrow;
    vrow += 32;
  }

  // Normalize and store fp32 output.
  float rinv[8];
  #pragma unroll
  for (int j = 0; j < 8; j++) rinv[j] = 1.0f / lst[j];
  float* orow = out + ((size_t)b * Sn + s0) * Dn;
  #pragma unroll
  for (int t = 0; t < 8; t++) {
    #pragma unroll
    for (int j = 0; j < 8; j++) {
      orow[(size_t)(j + hi * 8) * Dn + t * 16 + ln] = acc[t][j] * rinv[j];
    }
  }
}

// ---------------------------------------------------------------------------
// Launch
// ---------------------------------------------------------------------------
extern "C" void kernel_launch(void* const* d_in, const int* in_sizes, int n_in,
                              void* d_out, int out_size, void* d_ws, size_t ws_size,
                              hipStream_t stream) {
  const float* x  = (const float*)d_in[0];
  const float* Wq = (const float*)d_in[1];
  const float* bq = (const float*)d_in[2];
  const float* Wk = (const float*)d_in[3];
  const float* bk = (const float*)d_in[4];
  const float* Wv = (const float*)d_in[5];
  const float* bv = (const float*)d_in[6];
  float* out = (float*)d_out;

  // Workspace layout (bytes):
  //   Wt : 384*2048*2          = 1,572,864
  //   Qb : 8*4096*128*2        = 8,388,608
  //   Kb : 8,388,608
  //   Vt : 8,388,608           (transposed [B][D][S])
  char* ws = (char*)d_ws;
  __bf16* Wt = (__bf16*)(ws);
  __bf16* Qb = (__bf16*)(ws + 1572864);
  __bf16* Kb = (__bf16*)(ws + 1572864 + 8388608);
  __bf16* Vt = (__bf16*)(ws + 1572864 + 2 * 8388608);

  // 384*2048 elements / 256 threads
  prep_weights<<<3072, 256, 0, stream>>>(Wq, Wk, Wv, Wt);
  // 2048 row-blocks * 3 matrices = 6144 wave-tasks / 4 waves per block
  qkv_proj<<<1536, 128, 0, stream>>>(x, Wt, bq, bk, bv, Qb, Kb, Vt);
  // 2048 query tiles / 4 waves per block
  flash_attn<<<512, 128, 0, stream>>>(Qb, Kb, Vt, out);
}
